// STTSRNet_53060025975285
// MI455X (gfx1250) — matrix operand order repaired and checked by reading, not compiled, not run on hardware
//
#include <hip/hip_runtime.h>
#include <hip/hip_bf16.h>
#include <math.h>

// ---------------------------------------------------------------------------
// CDNA5 WMMA (wave32): D(16x16 f32) = A(16x32 bf16) * B(32x16 bf16) + C
// ---------------------------------------------------------------------------
typedef __attribute__((ext_vector_type(16))) __bf16 v16bf;
typedef __attribute__((ext_vector_type(8)))  float  v8f;

union Frag16 {
  uint4          q[2];
  unsigned short u[16];
  v16bf          v;
};

__device__ __forceinline__ unsigned short f2bf(float f) {
  __hip_bfloat16 h = __float2bfloat16(f);          // hardware cvt path
  return __builtin_bit_cast(unsigned short, h);
}

static inline int ceil_div(int a, int b) { return (a + b - 1) / b; }

// ---------------------------------------------------------------------------
// Weight pack: OIHW fp32 -> bf16 [Cout][tap][Cin]  (A-rows contiguous per tap)
// ---------------------------------------------------------------------------
__global__ void pack_w_k(const float* __restrict__ w, unsigned short* __restrict__ o,
                         int Cin, int Cout) {
  const int n = Cout * Cin * 9;
  int i = blockIdx.x * blockDim.x + threadIdx.x;
  if (i >= n) return;
  const int m   = i / (Cin * 9);
  const int rem = i - m * (Cin * 9);
  const int t   = rem / Cin;
  const int ci  = rem - t * Cin;
  o[i] = f2bf(w[(m * Cin + ci) * 9 + t]);
}

// ---------------------------------------------------------------------------
// Input staging: NCHW fp32 (+optional residual) -> channel-last bf16 [b][pix][Cin]
// ---------------------------------------------------------------------------
__global__ void to_cl_bf16_k(const float* __restrict__ x, const float* __restrict__ res,
                             unsigned short* __restrict__ o, int Cin, int HW) {
  const long long n = 4LL * HW * Cin;
  long long i = (long long)blockIdx.x * blockDim.x + threadIdx.x;
  if (i >= n) return;
  const int c = (int)(i % Cin);
  const int p = (int)((i / Cin) % HW);
  const int b = (int)(i / ((long long)Cin * HW));
  const size_t idx = ((size_t)b * Cin + c) * HW + p;
  float v = x[idx];
  if (res) v += res[idx];
  o[i] = f2bf(v);
}

// ---------------------------------------------------------------------------
// Implicit-GEMM 3x3 SAME conv.  One wave: 16 pixels x (MT*16) output channels.
// B fragment = two b128 loads from channel-last bf16 input; reused MT times.
// grid = (ceil(HW/16), 1, 4), block = 32.
// ---------------------------------------------------------------------------
template <int MT, int CIN>
__global__ void __launch_bounds__(32)
conv3x3_wmma_k(const unsigned short* __restrict__ xcl, const unsigned short* __restrict__ wp,
               const float* __restrict__ bias, float* __restrict__ out,
               int H, int W, int act, int cl_out) {
  const int Cout = MT * 16;
  const int HW = H * W;
  const int K9 = CIN * 9;
  const int l = threadIdx.x, lm = l & 15, lg = l >> 4;
  const int b = blockIdx.z;
  const int p = blockIdx.x * 16 + lm;
  const int py = p / W, px = p % W;
  const bool pok = (p < HW);

  v8f acc[MT];
#pragma unroll
  for (int m = 0; m < MT; ++m) acc[m] = {};

#pragma unroll
  for (int t = 0; t < 9; ++t) {
    const int dy = t / 3 - 1, dx = t % 3 - 1;     // compile-time after unroll
    const int yy = py + dy, xx = px + dx;
    const bool valid = pok && yy >= 0 && yy < H && xx >= 0 && xx < W;
    const unsigned short* bp = xcl + ((size_t)b * HW + yy * W + xx) * CIN;
#pragma unroll
    for (int kc = 0; kc < CIN / 32; ++kc) {
      Frag16 bb;
      if (valid) {   // per-lane EXEC-masked vector loads; WMMA itself is uniform
        bb.q[0] = *(const uint4*)(bp + kc * 32 + lg * 8);
        bb.q[1] = *(const uint4*)(bp + kc * 32 + 16 + lg * 8);
      } else {
        bb.q[0] = make_uint4(0u, 0u, 0u, 0u);
        bb.q[1] = make_uint4(0u, 0u, 0u, 0u);
      }
#pragma unroll
      for (int m = 0; m < MT; ++m) {
        const unsigned short* ap = wp + (size_t)(m * 16 + lm) * K9 + t * CIN + kc * 32;
        Frag16 a;
        a.q[0] = *(const uint4*)(ap + lg * 8);
        a.q[1] = *(const uint4*)(ap + 16 + lg * 8);
        acc[m] = __builtin_amdgcn_wmma_f32_16x16x32_bf16(false, a.v, false, bb.v,
                                                         (short)0, acc[m], false, false);
      }
    }
  }

#pragma unroll
  for (int m = 0; m < MT; ++m)
#pragma unroll
    for (int r = 0; r < 8; ++r) {
      const int co = m * 16 + r + lg * 8;         // C layout: VGPR r = rows {r, r+8}
      if (pok) {
        float o = acc[m][r] + bias[co];
        if (act == 1) o = fmaxf(o, 0.0f);
        if (cl_out) out[((size_t)b * HW + p) * Cout + co] = o;
        else        out[((size_t)b * Cout + co) * HW + p] = o;
      }
    }
}

// ---------------------------------------------------------------------------
// Scalar 3x3 conv for odd shapes (3->64 PReLU, 64->15). act:0 none,1 relu,2 prelu
// ---------------------------------------------------------------------------
__global__ void conv3x3_scalar_k(const float* __restrict__ x, const float* __restrict__ w,
                                 const float* __restrict__ bias, float* __restrict__ out,
                                 int Cin, int Cout, int H, int W, int act,
                                 const float* __restrict__ pa) {
  const int HW = H * W;
  const long long n = 4LL * Cout * HW;
  long long i = (long long)blockIdx.x * blockDim.x + threadIdx.x;
  if (i >= n) return;
  const int p  = (int)(i % HW);
  const int co = (int)((i / HW) % Cout);
  const int b  = (int)(i / ((long long)HW * Cout));
  const int y = p / W, xx = p % W;
  float acc = bias[co];
  for (int ci = 0; ci < Cin; ++ci) {
#pragma unroll
    for (int t = 0; t < 9; ++t) {
      const int yy = y + t / 3 - 1, xc = xx + t % 3 - 1;
      if (yy < 0 || yy >= H || xc < 0 || xc >= W) continue;
      acc += w[(co * Cin + ci) * 9 + t] * x[(((size_t)b * Cin + ci) * H + yy) * W + xc];
    }
  }
  if (act == 1) acc = fmaxf(acc, 0.0f);
  else if (act == 2) { const float a = pa[0]; acc = acc >= 0.0f ? acc : a * acc; }
  out[i] = acc;
}

// ---------------------------------------------------------------------------
// unfold3 + L2-normalize from channel-last fp32 [b][pix][32] -> bf16 [b][pix][288]
// Feature order [tap][c] (any fixed permutation is valid: dot & norm invariant).
// ---------------------------------------------------------------------------
__global__ void unfold_norm_cl_k(const float* __restrict__ qcl,
                                 unsigned short* __restrict__ Qn, int H, int W) {
  const int Cq = 32, HW = H * W;
  const long long n = 4LL * HW;
  long long i = (long long)blockIdx.x * blockDim.x + threadIdx.x;
  if (i >= n) return;
  const int p = (int)(i % HW);
  const int b = (int)(i / HW);
  const int y = p / W, xx = p % W;

  float ss = 0.0f;
#pragma unroll
  for (int t = 0; t < 9; ++t) {
    const int yy = y + t / 3 - 1, xc = xx + t % 3 - 1;
    if (yy < 0 || yy >= H || xc < 0 || xc >= W) continue;
    const float4* v4 = (const float4*)(qcl + ((size_t)b * HW + yy * W + xc) * Cq);
#pragma unroll
    for (int j = 0; j < 8; ++j) {
      const float4 v = v4[j];
      ss += v.x * v.x + v.y * v.y + v.z * v.z + v.w * v.w;
    }
  }
  const float inv = 1.0f / fmaxf(sqrtf(ss), 1e-12f);

  unsigned short* o = Qn + ((size_t)b * HW + p) * 288;
#pragma unroll
  for (int t = 0; t < 9; ++t) {
    const int yy = y + t / 3 - 1, xc = xx + t % 3 - 1;
    const bool valid = (yy >= 0 && yy < H && xc >= 0 && xc < W);
    const float4* v4 = (const float4*)(qcl + ((size_t)b * HW + yy * W + xc) * Cq);
#pragma unroll
    for (int j = 0; j < 8; ++j) {
      const float4 v = valid ? v4[j] : make_float4(0.f, 0.f, 0.f, 0.f);
      o[t * 32 + j * 4 + 0] = f2bf(v.x * inv);
      o[t * 32 + j * 4 + 1] = f2bf(v.y * inv);
      o[t * 32 + j * 4 + 2] = f2bf(v.z * inv);
      o[t * 32 + j * 4 + 3] = f2bf(v.w * inv);
    }
  }
}

// ---------------------------------------------------------------------------
// Flash-style correlation max/argmax.  Q block (64x288 bf16 = 36KB) resident in
// LDS for all HW/16 K-tile iterations; K tile (16x288 = 9KB) streamed per iter.
// ---------------------------------------------------------------------------
__global__ void __launch_bounds__(128)
attn_argmax_k(const unsigned short* __restrict__ Qn, const unsigned short* __restrict__ Kn,
              float* __restrict__ Rmax, int* __restrict__ Ridx, int HW) {
  const int K = 288;
  __shared__ __align__(16) unsigned short sQ[64 * 288];
  __shared__ __align__(16) unsigned short sK[16 * 288];

  const int tid = threadIdx.x;
  const int b   = blockIdx.y;
  const int q0  = blockIdx.x * 64;

  {  // stage Q block once
    const uint4* g = (const uint4*)(Qn + ((size_t)b * HW + q0) * K);
    uint4* s = (uint4*)sQ;
    for (int i = tid; i < 64 * K / 8; i += 128) s[i] = g[i];
  }

  const int w  = tid >> 5;
  const int l  = tid & 31;
  const int lm = l & 15;
  const int lg = l >> 4;

  float best = -3.4e38f;
  int   bidx = 0;
  const int nkt = HW >> 4;

  for (int kt = 0; kt < nkt; ++kt) {
    {  // stage K tile
      const uint4* g = (const uint4*)(Kn + ((size_t)b * HW + kt * 16) * K);
      uint4* s = (uint4*)sK;
      for (int i = tid; i < 16 * K / 8; i += 128) s[i] = g[i];
    }
    if (kt + 1 < nkt)  // prefetch next K tile (global_prefetch_b8), 72B/thread
      __builtin_prefetch((const void*)(Kn + ((size_t)b * HW + (kt + 1) * 16) * K + tid * 36), 0, 0);
    __syncthreads();

    v8f acc = {};
#pragma unroll
    for (int kc = 0; kc < 9; ++kc) {
      Frag16 a, bb;
      a.q[0]  = *(const uint4*)(sK + lm * K + kc * 32 + lg * 8);
      a.q[1]  = *(const uint4*)(sK + lm * K + kc * 32 + 16 + lg * 8);
      const int qb = (w * 16 + lm) * K + kc * 32 + lg * 8;
      bb.q[0] = *(const uint4*)(sQ + qb);
      bb.q[1] = *(const uint4*)(sQ + qb + 16);
      acc = __builtin_amdgcn_wmma_f32_16x16x32_bf16(false, a.v, false, bb.v,
                                                    (short)0, acc, false, false);
    }
#pragma unroll
    for (int r = 0; r < 8; ++r) {
      const float v = acc[r];
      const int kp = kt * 16 + r + lg * 8;
      if (v > best) { best = v; bidx = kp; }
    }
    __syncthreads();
  }

  const float ob = __shfl_xor(best, 16);
  const int   oi = __shfl_xor(bidx, 16);
  if (ob > best) { best = ob; bidx = oi; }
  if (lg == 0) {
    const int qq = q0 + w * 16 + lm;
    Rmax[(size_t)b * HW + qq] = best;
    Ridx[(size_t)b * HW + qq] = bidx;
  }
}

// ---------------------------------------------------------------------------
// gather(v-patches via Ridx) + fold + /9, times S, plus feat + x
// ---------------------------------------------------------------------------
__global__ void fold_combine_k(const float* __restrict__ feat, const float* __restrict__ xin,
                               const float* __restrict__ v, const float* __restrict__ Rmax,
                               const int* __restrict__ Ridx, float* __restrict__ out,
                               int C, int H, int W) {
  const int HW = H * W;
  const long long n = 4LL * C * HW;
  long long i = (long long)blockIdx.x * blockDim.x + threadIdx.x;
  if (i >= n) return;
  const int p = (int)(i % HW);
  const int c = (int)((i / HW) % C);
  const int b = (int)(i / ((long long)HW * C));
  const int y = p / W, xx = p % W;
  float acc = 0.0f;
#pragma unroll
  for (int ki = 0; ki < 3; ++ki)
#pragma unroll
    for (int kj = 0; kj < 3; ++kj) {
      const int qy = y + 1 - ki, qx = xx + 1 - kj;
      if (qy < 0 || qy >= H || qx < 0 || qx >= W) continue;
      const int qp = qy * W + qx;
      const int kp = Ridx[(size_t)b * HW + qp];
      const int ky = kp / W + ki - 1, kx = kp % W + kj - 1;
      if (ky >= 0 && ky < H && kx >= 0 && kx < W)
        acc += v[(((size_t)b * C + c) * H + ky) * W + kx];
    }
  out[i] = feat[i] + xin[i] + (acc * (1.0f / 9.0f)) * Rmax[(size_t)b * HW + p];
}

// ---------------------------------------------------------------------------
// pool / upsample / concat
// ---------------------------------------------------------------------------
__global__ void pool2x2_k(const float* __restrict__ in, float* __restrict__ out,
                          int C, int Ho, int Wo) {
  const long long n = 4LL * C * Ho * Wo;
  long long i = (long long)blockIdx.x * blockDim.x + threadIdx.x;
  if (i >= n) return;
  const int xo = (int)(i % Wo);
  const int yo = (int)((i / Wo) % Ho);
  const int c  = (int)((i / ((long long)Wo * Ho)) % C);
  const int b  = (int)(i / ((long long)Wo * Ho * C));
  const int Wi = 2 * Wo;
  const float* p = in + (((size_t)b * C + c) * (2 * Ho) + 2 * yo) * Wi + 2 * xo;
  out[i] = 0.25f * (p[0] + p[1] + p[Wi] + p[Wi + 1]);
}

__global__ void upsample2_k(const float* __restrict__ in, float* __restrict__ out,
                            int C, int Ho, int Wo) {
  const long long n = 4LL * C * Ho * Wo;
  long long i = (long long)blockIdx.x * blockDim.x + threadIdx.x;
  if (i >= n) return;
  const int xo = (int)(i % Wo);
  const int yo = (int)((i / Wo) % Ho);
  const int c  = (int)((i / ((long long)Wo * Ho)) % C);
  const int b  = (int)(i / ((long long)Wo * Ho * C));
  out[i] = in[(((size_t)b * C + c) * (Ho / 2) + yo / 2) * (Wo / 2) + xo / 2];
}

__global__ void concat2_k(const float* __restrict__ a, const float* __restrict__ bb,
                          float* __restrict__ out, int H, int W) {
  const int HW = H * W;
  const long long n = 4LL * 128 * HW;
  long long i = (long long)blockIdx.x * blockDim.x + threadIdx.x;
  if (i >= n) return;
  const int p = (int)(i % HW);
  const int c = (int)((i / HW) % 128);
  const int b = (int)(i / ((long long)HW * 128));
  out[i] = (c < 64) ? a[((size_t)b * 64 + c) * HW + p]
                    : bb[((size_t)b * 64 + (c - 64)) * HW + p];
}

__global__ void concat_up_k(const float* __restrict__ a, const float* __restrict__ bsm,
                            float* __restrict__ out, int H, int W) {
  const int HW = H * W;
  const long long n = 4LL * 128 * HW;
  long long i = (long long)blockIdx.x * blockDim.x + threadIdx.x;
  if (i >= n) return;
  const int p = (int)(i % HW);
  const int c = (int)((i / HW) % 128);
  const int b = (int)(i / ((long long)HW * 128));
  if (c < 64) {
    out[i] = a[((size_t)b * 64 + c) * HW + p];
  } else {
    const int y = p / W, xx = p % W;
    out[i] = bsm[(((size_t)b * 64 + (c - 64)) * (H / 2) + y / 2) * (W / 2) + xx / 2];
  }
}

// ---------------------------------------------------------------------------
// Host-side orchestration
// ---------------------------------------------------------------------------
struct ConvP { const float* w; const float* b; };
struct TbP  { ConvP f1, f2, k, q, v1, v2; };

static ConvP convp(void* const* din, int ib) {
  ConvP p; p.b = (const float*)din[ib]; p.w = (const float*)din[ib + 1]; return p;
}
static TbP tbp(void* const* din, int i0) {
  TbP t;
  t.f1 = convp(din, i0);      t.f2 = convp(din, i0 + 2);
  t.k  = convp(din, i0 + 4);  t.q  = convp(din, i0 + 6);
  t.v1 = convp(din, i0 + 8);  t.v2 = convp(din, i0 + 10);
  return t;
}

struct Ws {
  unsigned short *pack, *xcl, *Qn, *Kn;
  float *h, *down1, *pool1, *down2, *pool2, *down3, *pool3, *bott, *f1c, *pre1,
        *up1, *f2c, *pre2, *up2, *f3c, *pre3, *up3, *i3, *o1out;
  float *qb, *kb, *t1, *t2, *t3, *t4, *Rmax;
  int* Ridx;
};

static void conv_wmma(hipStream_t st, const float* x, const ConvP& p, const float* res,
                      float* out, int Cin, int Cout, int H, int W, int act, int cl_out,
                      Ws& s) {
  const int HW = H * W;
  const long long ncl = 4LL * HW * Cin;
  to_cl_bf16_k<<<(unsigned)((ncl + 255) / 256), 256, 0, st>>>(x, res, s.xcl, Cin, HW);
  const int nw = Cout * Cin * 9;
  pack_w_k<<<ceil_div(nw, 256), 256, 0, st>>>(p.w, s.pack, Cin, Cout);
  dim3 g(ceil_div(HW, 16), 1, 4);
  if (Cin == 64 && Cout == 64)
    conv3x3_wmma_k<4, 64><<<g, 32, 0, st>>>(s.xcl, s.pack, p.b, out, H, W, act, cl_out);
  else if (Cin == 64 && Cout == 32)
    conv3x3_wmma_k<2, 64><<<g, 32, 0, st>>>(s.xcl, s.pack, p.b, out, H, W, act, cl_out);
  else  // Cin == 128, Cout == 64
    conv3x3_wmma_k<4, 128><<<g, 32, 0, st>>>(s.xcl, s.pack, p.b, out, H, W, act, cl_out);
}

static void conv_scalar(hipStream_t st, const float* x, const ConvP& p, float* out,
                        int Cin, int Cout, int H, int W, int act, const float* pa) {
  const long long n = 4LL * Cout * H * W;
  conv3x3_scalar_k<<<(unsigned)((n + 255) / 256), 256, 0, st>>>(x, p.w, p.b, out,
                                                                Cin, Cout, H, W, act, pa);
}

static void run_tblock(hipStream_t st, const float* x, const TbP& p, float* out,
                       int H, int W, Ws& s) {
  const int HW = H * W;
  // v = conv(conv(x, v1) + x, v2)                (NCHW outputs)
  conv_wmma(st, x, p.v1, nullptr, s.t1, 64, 64, H, W, 0, 0, s);
  conv_wmma(st, s.t1, p.v2, x, s.t2, 64, 64, H, W, 0, 0, s);
  // q, k (64 -> 32), channel-last outputs for vectorized unfold+normalize
  conv_wmma(st, x, p.q, nullptr, s.qb, 64, 32, H, W, 0, 1, s);
  conv_wmma(st, x, p.k, nullptr, s.kb, 64, 32, H, W, 0, 1, s);
  const int np = 4 * HW;
  unfold_norm_cl_k<<<ceil_div(np, 128), 128, 0, st>>>(s.qb, s.Qn, H, W);
  unfold_norm_cl_k<<<ceil_div(np, 128), 128, 0, st>>>(s.kb, s.Kn, H, W);
  attn_argmax_k<<<dim3(HW / 64, 4), 128, 0, st>>>(s.Qn, s.Kn, s.Rmax, s.Ridx, HW);
  // feat = relu(conv(relu(conv(x, f1)), f2))
  conv_wmma(st, x, p.f1, nullptr, s.t3, 64, 64, H, W, 1, 0, s);
  conv_wmma(st, s.t3, p.f2, nullptr, s.t4, 64, 64, H, W, 1, 0, s);
  const long long n2 = 4LL * 64 * HW;
  fold_combine_k<<<(unsigned)((n2 + 255) / 256), 256, 0, st>>>(s.t4, x, s.t2, s.Rmax,
                                                               s.Ridx, out, 64, H, W);
}

extern "C" void kernel_launch(void* const* d_in, const int* in_sizes, int n_in,
                              void* d_out, int out_size, void* d_ws, size_t ws_size,
                              hipStream_t stream) {
  (void)in_sizes; (void)n_in; (void)out_size; (void)ws_size;

  // jax pytree flatten order (dict keys sorted alphabetically):
  // 0 bottom.b 1 bottom.w | 2..13 d1 | 14..25 d2 | 26..37 d3 | 38 in_conv.b 39 in_conv.w
  // 40 o1.b 41 o1.w | 42 o2.b 43 o2.w | 44 prelu_a | 45..56 u1 | 57 u1pre.b 58 u1pre.w
  // 59..70 u2 | 71 u2pre.b 72 u2pre.w | 73..84 u3 | 85 u3pre.b 86 u3pre.w | 87 x
  const float* x       = (const float*)d_in[87];
  const float* prelu_a = (const float*)d_in[44];
  ConvP in_conv = convp(d_in, 38), bottom = convp(d_in, 0);
  ConvP o1 = convp(d_in, 40), o2 = convp(d_in, 42);
  ConvP u1pre = convp(d_in, 57), u2pre = convp(d_in, 71), u3pre = convp(d_in, 85);
  TbP d1 = tbp(d_in, 2), d2 = tbp(d_in, 14), d3 = tbp(d_in, 26);
  TbP u1 = tbp(d_in, 45), u2 = tbp(d_in, 59), u3 = tbp(d_in, 73);

  // workspace carve (bump allocator, 256B aligned)
  char* wsb = (char*)d_ws;
  size_t off = 0;
  auto A = [&](size_t bytes) -> void* {
    size_t o = (off + 255) & ~(size_t)255; off = o + bytes; return wsb + o;
  };
  Ws s;
  const size_t L1 = 4ull * 64 * 4096 * 4;     // 4 MB  (B,64,64,64) f32
  s.pack  = (unsigned short*)A(64ull * 1152 * 2);
  s.xcl   = (unsigned short*)A(4ull * 4096 * 128 * 2);
  s.h     = (float*)A(L1);      s.down1 = (float*)A(L1);
  s.pool1 = (float*)A(L1 / 4);  s.down2 = (float*)A(L1 / 4);
  s.pool2 = (float*)A(L1 / 16); s.down3 = (float*)A(L1 / 16);
  s.pool3 = (float*)A(L1 / 64); s.bott  = (float*)A(L1 / 64);
  s.f1c   = (float*)A(L1 / 32); s.pre1  = (float*)A(L1 / 64); s.up1 = (float*)A(L1 / 64);
  s.f2c   = (float*)A(L1 / 8);  s.pre2  = (float*)A(L1 / 16); s.up2 = (float*)A(L1 / 16);
  s.f3c   = (float*)A(L1 / 2);  s.pre3  = (float*)A(L1 / 4);  s.up3 = (float*)A(L1 / 4);
  s.i3    = (float*)A(L1);      s.o1out = (float*)A(L1);
  s.qb    = (float*)A(L1 / 2);  s.kb    = (float*)A(L1 / 2);
  s.t1 = (float*)A(L1); s.t2 = (float*)A(L1); s.t3 = (float*)A(L1); s.t4 = (float*)A(L1);
  s.Rmax  = (float*)A(4ull * 4096 * 4);
  s.Ridx  = (int*)  A(4ull * 4096 * 4);
  s.Qn    = (unsigned short*)A(4ull * 4096 * 288 * 2);
  s.Kn    = (unsigned short*)A(4ull * 4096 * 288 * 2);

  // ----- network -----
  conv_scalar(stream, x, in_conv, s.h, 3, 64, 64, 64, 2, prelu_a);        // PReLU

  run_tblock(stream, s.h, d1, s.down1, 64, 64, s);
  pool2x2_k<<<ceil_div(4 * 64 * 32 * 32, 256), 256, 0, stream>>>(s.down1, s.pool1, 64, 32, 32);
  run_tblock(stream, s.pool1, d2, s.down2, 32, 32, s);
  pool2x2_k<<<ceil_div(4 * 64 * 16 * 16, 256), 256, 0, stream>>>(s.down2, s.pool2, 64, 16, 16);
  run_tblock(stream, s.pool2, d3, s.down3, 16, 16, s);
  pool2x2_k<<<ceil_div(4 * 64 * 8 * 8, 256), 256, 0, stream>>>(s.down3, s.pool3, 64, 8, 8);

  conv_wmma(stream, s.pool3, bottom, nullptr, s.bott, 64, 64, 8, 8, 0, 0, s);
  concat2_k<<<ceil_div(4 * 128 * 8 * 8, 256), 256, 0, stream>>>(s.pool3, s.bott, s.f1c, 8, 8);
  conv_wmma(stream, s.f1c, u1pre, nullptr, s.pre1, 128, 64, 8, 8, 1, 0, s);
  run_tblock(stream, s.pre1, u1, s.up1, 8, 8, s);

  concat_up_k<<<ceil_div(4 * 128 * 16 * 16, 256), 256, 0, stream>>>(s.pool2, s.up1, s.f2c, 16, 16);
  conv_wmma(stream, s.f2c, u2pre, nullptr, s.pre2, 128, 64, 16, 16, 1, 0, s);
  run_tblock(stream, s.pre2, u2, s.up2, 16, 16, s);

  concat_up_k<<<ceil_div(4 * 128 * 32 * 32, 256), 256, 0, stream>>>(s.pool1, s.up2, s.f3c, 32, 32);
  conv_wmma(stream, s.f3c, u3pre, nullptr, s.pre3, 128, 64, 32, 32, 1, 0, s);
  run_tblock(stream, s.pre3, u3, s.up3, 32, 32, s);

  upsample2_k<<<ceil_div(4 * 64 * 64 * 64, 256), 256, 0, stream>>>(s.up3, s.i3, 64, 64, 64);
  conv_wmma(stream, s.i3, o1, nullptr, s.o1out, 64, 64, 64, 64, 1, 0, s);
  conv_scalar(stream, s.o1out, o2, (float*)d_out, 64, 15, 64, 64, 0, nullptr);
}
